// ESTAG_50766513438889
// MI455X (gfx1250) — matrix operand order repaired and checked by reading, not compiled, not run on hardware
//
#include <hip/hip_runtime.h>
#include <hip/hip_bf16.h>
#include <math.h>

// ---------------- problem constants ----------------
constexpr int kP  = 10;      // past frames
constexpr int kD  = 64;      // feature dim
constexpr int kBN = 10000;   // nodes
constexpr int kE  = 160000;  // edges
constexpr int kEA = 9;       // P-1 edge attrs
constexpr int kNL = 2;       // layers
constexpr int kKE = 160;     // edge MLP K padded (138 -> 160, 5 k-steps of 32)
constexpr int kKEr = 138;    // real edge MLP K = 2D+1+EA
constexpr int kKN = 128;     // node MLP K = 2D

typedef __attribute__((ext_vector_type(16))) _Float16 v16h;
typedef __attribute__((ext_vector_type(8)))  _Float16 v8h;
typedef __attribute__((ext_vector_type(8)))  float    v8f;

// fast SiLU: x * rcp(1+exp(-x)) -> v_exp_f32 + v_add + v_rcp_f32 + v_mul
// (avoids the IEEE v_div_scale/v_div_fmas sequence; activation precision is ample)
__device__ __forceinline__ float silu_f(float x) {
  return x * __builtin_amdgcn_rcpf(1.0f + __expf(-x));
}

// ---------------- WMMA fragment helpers (ISA 7.12.2 layouts, wave32) ----------------
// A: 16x32 f16 tile from LDS row-major [16 x lda] via two 16B vector loads per lane.
// lanes 0-15: M=lane, halves = K[k0+0..7 | k0+16..23]; lanes 16-31: M=lane-16, +8.
__device__ __forceinline__ v16h load_a_frag(const _Float16* A, int lda, int k0, int lane) {
  int r  = lane & 15;
  int kb = (lane >> 4) << 3;   // 0 or 8
  v8h lo = *(const v8h*)(A + r * lda + k0 + kb);
  v8h hi = *(const v8h*)(A + r * lda + k0 + 16 + kb);
  return __builtin_shufflevector(lo, hi, 0, 1, 2, 3, 4, 5, 6, 7,
                                 8, 9, 10, 11, 12, 13, 14, 15);
}

// B kept in LDS pre-packed fragment-major: for (tile, chunk, lane) the 16 halves a
// lane consumes are contiguous (32B) -> one v16h load (2x ds_load_b128), no packing VALU.
// Bp[(((n0/16)*(K/32) + k0/32)*32 + lane)*16 + e] = W[k, n] with
//   k = (k0/32)*32 + (e>>3)*16 + (lane>>4)*8 + (e&7),  n = (n0/16)*16 + (lane&15)
__device__ __forceinline__ void pack_b_lds(_Float16* Bp, const float* __restrict__ Wg,
                                           int K, int Kreal, int tid) {
  const int nch = K >> 5;
  const int total = K * kD;               // 4 tiles * nch * 32 lanes * 16 halves
  for (int i = tid; i < total; i += 128) {
    int e    = i & 15;
    int lane = (i >> 4) & 31;
    int blk  = i >> 9;                    // (tile * nch + ch)
    int ch   = blk % nch;
    int tile = blk / nch;
    int k = (ch << 5) + ((e >> 3) << 4) + ((lane >> 4) << 3) + (e & 7);
    int n = (tile << 4) + (lane & 15);
    Bp[i] = (_Float16)((k < Kreal) ? Wg[k * kD + n] : 0.0f);
  }
}

__device__ __forceinline__ v16h load_b_packed(const _Float16* Bp, int K, int n0, int k0,
                                              int lane) {
  return *(const v16h*)(Bp + ((((n0 >> 4) * (K >> 5)) + (k0 >> 5)) * 32 + lane) * 16);
}

// 16xK @ Kx16 tile via v_wmma_f32_16x16x32_f16 accumulation (B pre-packed)
__device__ __forceinline__ v8f wmma_tile(const _Float16* A, int lda,
                                         const _Float16* Bp, int K, int n0, int lane) {
  v8f acc = {};
  for (int k0 = 0; k0 < K; k0 += 32) {
    v16h a = load_a_frag(A, lda, k0, lane);
    v16h b = load_b_packed(Bp, K, n0, k0, lane);
    acc = __builtin_amdgcn_wmma_f32_16x16x32_f16(false, a, false, b, (short)0, acc, false, false);
  }
  return acc;
}

// C/D layout: lane&15 = N, (lane>>4)*8 + i = M.
__device__ __forceinline__ void store_tile_lds(v8f acc, _Float16* Out, int ldo, int n0,
                                               const float* bias, int lane, bool do_silu) {
  int n  = (lane & 15) + n0;
  int mb = (lane >> 4) << 3;
#pragma unroll
  for (int i = 0; i < 8; ++i) {
    float xv = acc[i] + bias[n];
    if (do_silu) xv = silu_f(xv);
    Out[(mb + i) * ldo + n] = (_Float16)xv;
  }
}

// ---------------- small utility kernels ----------------
__global__ void k_copy(const float* __restrict__ s, float* __restrict__ d, int n) {
  int i = blockIdx.x * blockDim.x + threadIdx.x;
  if (i < n) d[i] = s[i];
}
__global__ void k_zero(float* __restrict__ p, int n) {
  int i = blockIdx.x * blockDim.x + threadIdx.x;
  if (i < n) p[i] = 0.0f;
}

// centered DFT over P frames, nonzero freqs 1..P-1 -> Fq[n][f][c] = (re,im)
__global__ void k_dft(const float* __restrict__ x, float* __restrict__ Fq) {
  int n = blockIdx.x * blockDim.x + threadIdx.x;
  if (n >= kBN) return;
  float xc[kP][3];
  float mc[3] = {0.f, 0.f, 0.f};
#pragma unroll
  for (int p = 0; p < kP; ++p)
#pragma unroll
    for (int c = 0; c < 3; ++c) {
      float v = x[(n * kP + p) * 3 + c];
      xc[p][c] = v; mc[c] += v;
    }
#pragma unroll
  for (int c = 0; c < 3; ++c) mc[c] *= (1.0f / kP);
  for (int f = 0; f < kEA; ++f) {
#pragma unroll
    for (int c = 0; c < 3; ++c) {
      float re = 0.f, im = 0.f;
#pragma unroll
      for (int p = 0; p < kP; ++p) {
        float ang = -0.6283185307179586f * (float)((f + 1) * p); // -2*pi/P * fp
        float v = xc[p][c] - mc[c];
        re += v * __cosf(ang);
        im += v * __sinf(ang);
      }
      Fq[((size_t)n * kEA + f) * 6 + c * 2 + 0] = re;
      Fq[((size_t)n * kEA + f) * 6 + c * 2 + 1] = im;
    }
  }
}

// edge_attr[e][f] = | sum_c Fq[row][f][c] * conj(Fq[col][f][c]) |
__global__ void k_edge_attr(const float* __restrict__ Fq, const int* __restrict__ row,
                            const int* __restrict__ col, float* __restrict__ eat) {
  int e = blockIdx.x * blockDim.x + threadIdx.x;
  if (e >= kE) return;
  int r = row[e], c = col[e];
  for (int f = 0; f < kEA; ++f) {
    float re = 0.f, im = 0.f;
#pragma unroll
    for (int cc = 0; cc < 3; ++cc) {
      float ar = Fq[((size_t)r * kEA + f) * 6 + cc * 2 + 0];
      float ai = Fq[((size_t)r * kEA + f) * 6 + cc * 2 + 1];
      float br = Fq[((size_t)c * kEA + f) * 6 + cc * 2 + 0];
      float bi = Fq[((size_t)c * kEA + f) * 6 + cc * 2 + 1];
      re += ar * br + ai * bi;
      im += ai * br - ar * bi;
    }
    eat[(size_t)e * kEA + f] = sqrtf(re * re + im * im);
  }
}

__global__ void k_cnt(const int* __restrict__ row, float* __restrict__ cnt) {
  int e = blockIdx.x * blockDim.x + threadIdx.x;
  if (e < kE) atomicAdd(&cnt[row[e]], 1.0f);
}

// hh = h @ embW + embB + time_emb
__global__ void k_embed(const float* __restrict__ h, const float* __restrict__ embW,
                        const float* __restrict__ embB, const float* __restrict__ temb,
                        float* __restrict__ hh) {
  int i = blockIdx.x * blockDim.x + threadIdx.x;
  if (i >= kBN * kP * kD) return;
  int d = i % kD;
  int p = (i / kD) % kP;
  int np = i / kD; // n*kP + p
  hh[i] = h[np] * embW[d] + embB[d] + temb[p * kD + d];
}

// ---------------- fused edge MLP + coord MLP (per layer, per t) ----------------
__global__ __launch_bounds__(128)
void k_edge_fused(const float* __restrict__ hh, const float* __restrict__ xx,
                  const int* __restrict__ row, const int* __restrict__ col,
                  const float* __restrict__ eat,
                  const float* __restrict__ W1, const float* __restrict__ b1,
                  const float* __restrict__ W2, const float* __restrict__ b2,
                  const float* __restrict__ Wc1, const float* __restrict__ bc1,
                  const float* __restrict__ wc2,
                  float* __restrict__ agg, float* __restrict__ xacc, int t) {
  __shared__ __align__(32) _Float16 As[16 * kKE];
  __shared__ __align__(32) _Float16 W1s[kKE * kD];   // packed fragment-major
  __shared__ __align__(32) _Float16 W2s[kD * kD];    // packed
  __shared__ __align__(32) _Float16 Wc1s[kD * kD];   // packed
  __shared__ __align__(32) _Float16 Hs[16 * kD];
  __shared__ __align__(32) _Float16 EFs[16 * kD];
  __shared__ __align__(32) _Float16 Hcs[16 * kD];
  __shared__ float b1s[kD], b2s[kD], bc1s[kD], wc2s[kD];
  __shared__ float dn[16][3];
  __shared__ float rad[16];
  __shared__ int rn[16], cn[16];

  const int tid = threadIdx.x;
  const int lane = tid & 31;
  const int wv = tid >> 5;
  const int e0 = blockIdx.x * 16;

  // per-edge geometry
  if (tid < 16) {
    int e = e0 + tid;
    int r = row[e], c = col[e];
    rn[tid] = r; cn[tid] = c;
    float d0 = xx[((size_t)r * kP + t) * 3 + 0] - xx[((size_t)c * kP + t) * 3 + 0];
    float d1 = xx[((size_t)r * kP + t) * 3 + 1] - xx[((size_t)c * kP + t) * 3 + 1];
    float d2 = xx[((size_t)r * kP + t) * 3 + 2] - xx[((size_t)c * kP + t) * 3 + 2];
    float rr = d0 * d0 + d1 * d1 + d2 * d2;
    rad[tid] = rr;
    float nm = fmaxf(sqrtf(rr), 1e-12f);
    dn[tid][0] = d0 / nm; dn[tid][1] = d1 / nm; dn[tid][2] = d2 / nm;
  }
  // weights -> LDS, packed into B-fragment order (f32 -> f16)
  pack_b_lds(W1s, W1, kKE, kKEr, tid);
  pack_b_lds(W2s, W2, kD, kD, tid);
  pack_b_lds(Wc1s, Wc1, kD, kD, tid);
  for (int i = tid; i < kD; i += 128) {
    b1s[i] = b1[i]; b2s[i] = b2[i]; bc1s[i] = bc1[i]; wc2s[i] = wc2[i];
  }
  __syncthreads();

  // assemble A: [h[row] | h[col] | radial | edge_attr | 0-pad]
  for (int i = tid; i < 16 * kKE; i += 128) {
    int m = i / kKE, kk = i % kKE;
    int e = e0 + m;
    float v;
    if (kk < kD)                 v = hh[((size_t)rn[m] * kP + t) * kD + kk];
    else if (kk < 2 * kD)        v = hh[((size_t)cn[m] * kP + t) * kD + (kk - kD)];
    else if (kk == 2 * kD)       v = rad[m];
    else if (kk < 2 * kD + 1 + kEA) v = eat[(size_t)e * kEA + (kk - 2 * kD - 1)];
    else                         v = 0.0f;
    As[m * kKE + kk] = (_Float16)v;
  }
  __syncthreads();

  const int n0 = wv * 16;
  // edge MLP layer 1 (K=160) + SiLU
  v8f acc = wmma_tile(As, kKE, W1s, kKE, n0, lane);
  store_tile_lds(acc, Hs, kD, n0, b1s, lane, true);
  __syncthreads();
  // edge MLP layer 2 (K=64) + SiLU -> ef ; atomic segment-sum into agg[row]
  acc = wmma_tile(Hs, kD, W2s, kD, n0, lane);
  {
    int n = (lane & 15) + n0;
    int mb = (lane >> 4) << 3;
#pragma unroll
    for (int i = 0; i < 8; ++i) {
      int m = mb + i;
      float ef = silu_f(acc[i] + b2s[n]);
      EFs[m * kD + n] = (_Float16)ef;
      atomicAdd(&agg[(size_t)rn[m] * kD + n], ef);
    }
  }
  __syncthreads();
  // coord MLP layer 1 (K=64) + SiLU
  acc = wmma_tile(EFs, kD, Wc1s, kD, n0, lane);
  store_tile_lds(acc, Hcs, kD, n0, bc1s, lane, true);
  __syncthreads();
  // coord MLP layer 2: per-edge scalar w = Hc . wc2 ; scatter clipped trans
  if (tid < 16) {
    float s = 0.f;
    for (int k = 0; k < kD; ++k) s += (float)Hcs[tid * kD + k] * wc2s[k];
#pragma unroll
    for (int c = 0; c < 3; ++c) {
      float tr = fminf(fmaxf(dn[tid][c] * s, -100.0f), 100.0f);
      atomicAdd(&xacc[(size_t)rn[tid] * 3 + c], tr);
    }
  }
}

// x_t += xacc / max(cnt,1)
__global__ void k_coord_fin(float* __restrict__ xx, const float* __restrict__ xacc,
                            const float* __restrict__ cnt, int t) {
  int i = blockIdx.x * blockDim.x + threadIdx.x;
  if (i >= kBN * 3) return;
  int n = i / 3, c = i % 3;
  xx[((size_t)n * kP + t) * 3 + c] += xacc[i] / fmaxf(cnt[n], 1.0f);
}

// ---------------- fused node MLP (residual, per t) ----------------
__global__ __launch_bounds__(128)
void k_node_fused(float* __restrict__ hh, const float* __restrict__ agg,
                  const float* __restrict__ W1, const float* __restrict__ b1,
                  const float* __restrict__ W2, const float* __restrict__ b2, int t) {
  __shared__ __align__(32) _Float16 As[16 * kKN];
  __shared__ __align__(32) _Float16 W1s[kKN * kD];   // packed
  __shared__ __align__(32) _Float16 W2s[kD * kD];    // packed
  __shared__ __align__(32) _Float16 Hs[16 * kD];
  __shared__ float b1s[kD], b2s[kD];

  const int tid = threadIdx.x;
  const int lane = tid & 31;
  const int wv = tid >> 5;
  const int m0 = blockIdx.x * 16;  // node base

  pack_b_lds(W1s, W1, kKN, kKN, tid);
  pack_b_lds(W2s, W2, kD, kD, tid);
  for (int i = tid; i < kD; i += 128) { b1s[i] = b1[i]; b2s[i] = b2[i]; }

  for (int i = tid; i < 16 * kKN; i += 128) {
    int m = i / kKN, kk = i % kKN;
    int n = m0 + m;
    float v = (kk < kD) ? hh[((size_t)n * kP + t) * kD + kk]
                        : agg[(size_t)n * kD + (kk - kD)];
    As[m * kKN + kk] = (_Float16)v;
  }
  __syncthreads();

  const int n0 = wv * 16;
  v8f acc = wmma_tile(As, kKN, W1s, kKN, n0, lane);
  store_tile_lds(acc, Hs, kD, n0, b1s, lane, true);
  __syncthreads();
  acc = wmma_tile(Hs, kD, W2s, kD, n0, lane);
  {
    int n = (lane & 15) + n0;
    int mb = (lane >> 4) << 3;
#pragma unroll
    for (int i = 0; i < 8; ++i) {
      int m = m0 + mb + i;
      hh[((size_t)m * kP + t) * kD + n] += acc[i] + b2s[n];  // residual, no SiLU
    }
  }
}

// ---------------- generic fused 2-layer SiLU MLP (q/k/v) ----------------
__global__ __launch_bounds__(128)
void k_mlp2(const float* __restrict__ in, float* __restrict__ out,
            const float* __restrict__ W1, const float* __restrict__ b1,
            const float* __restrict__ W2, const float* __restrict__ b2) {
  __shared__ __align__(32) _Float16 As[16 * kD];
  __shared__ __align__(32) _Float16 W1s[kD * kD];    // packed
  __shared__ __align__(32) _Float16 W2s[kD * kD];    // packed
  __shared__ __align__(32) _Float16 Hs[16 * kD];
  __shared__ float b1s[kD], b2s[kD];

  const int tid = threadIdx.x;
  const int lane = tid & 31;
  const int wv = tid >> 5;
  const int m0 = blockIdx.x * 16;  // row base in [0, BN*P)

  // prefetch next block's A rows while we stage weights (global_prefetch_b8)
  __builtin_prefetch(&in[(size_t)(m0 + 16) * kD + tid * 8], 0, 1);

  pack_b_lds(W1s, W1, kD, kD, tid);
  pack_b_lds(W2s, W2, kD, kD, tid);
  for (int i = tid; i < kD; i += 128) { b1s[i] = b1[i]; b2s[i] = b2[i]; }
  for (int i = tid; i < 16 * kD; i += 128) {
    int m = i / kD, kk = i % kD;
    As[i] = (_Float16)in[(size_t)(m0 + m) * kD + kk];
  }
  __syncthreads();

  const int n0 = wv * 16;
  v8f acc = wmma_tile(As, kD, W1s, kD, n0, lane);
  store_tile_lds(acc, Hs, kD, n0, b1s, lane, true);
  __syncthreads();
  acc = wmma_tile(Hs, kD, W2s, kD, n0, lane);
  {
    int n = (lane & 15) + n0;
    int mb = (lane >> 4) << 3;
#pragma unroll
    for (int i = 0; i < 8; ++i)
      out[(size_t)(m0 + mb + i) * kD + n] = silu_f(acc[i] + b2s[n]);
  }
}

// ---------------- temporal equivariant attention (per node) ----------------
__global__ __launch_bounds__(128)
void k_attn(float* __restrict__ hh, float* __restrict__ xx,
            const float* __restrict__ q, const float* __restrict__ k,
            const float* __restrict__ v,
            const float* __restrict__ Wc1, const float* __restrict__ bc1,
            const float* __restrict__ wc2) {
  __shared__ float ql[kP * kD], kl[kP * kD], vl[kP * kD];
  __shared__ float Wc1s[kD * kD];
  __shared__ float bc1s[kD], wc2s[kD];
  __shared__ float al[kP * kP];
  __shared__ float hid[kP * kD];
  __shared__ float cvl[kP];
  __shared__ float xl[kP][3];

  const int tid = threadIdx.x;
  const int n = blockIdx.x;
  const size_t base = (size_t)n * kP;

  for (int i = tid; i < kP * kD; i += 128) {
    ql[i] = q[base * kD + i];
    kl[i] = k[base * kD + i];
    vl[i] = v[base * kD + i];
  }
  for (int i = tid; i < kD * kD; i += 128) Wc1s[i] = Wc1[i];
  for (int i = tid; i < kD; i += 128) { bc1s[i] = bc1[i]; wc2s[i] = wc2[i]; }
  for (int i = tid; i < kP * 3; i += 128) xl[i / 3][i % 3] = xx[base * 3 + i];
  __syncthreads();

  // scores
  if (tid < kP * kP) {
    int i = tid / kP, j = tid % kP;
    float s = 0.f;
    for (int d = 0; d < kD; ++d) s += ql[i * kD + d] * kl[j * kD + d];
    al[tid] = s;
  }
  __syncthreads();
  // row softmax
  if (tid < kP) {
    float mx = -1e30f;
    for (int j = 0; j < kP; ++j) mx = fmaxf(mx, al[tid * kP + j]);
    float sm = 0.f;
    for (int j = 0; j < kP; ++j) { float e = __expf(al[tid * kP + j] - mx); al[tid * kP + j] = e; sm += e; }
    float inv = __builtin_amdgcn_rcpf(sm);
    for (int j = 0; j < kP; ++j) al[tid * kP + j] *= inv;
  }
  __syncthreads();
  // h += alpha @ v ; hidden = silu(v @ Wc1 + bc1)
  for (int idx = tid; idx < kP * kD; idx += 128) {
    int i = idx / kD, d = idx % kD;
    float s = 0.f;
    for (int j = 0; j < kP; ++j) s += al[i * kP + j] * vl[j * kD + d];
    hh[(base + i) * kD + d] += s;
    float hsum = bc1s[d];
    for (int dd = 0; dd < kD; ++dd) hsum += vl[i * kD + dd] * Wc1s[dd * kD + d];
    hid[idx] = silu_f(hsum);
  }
  __syncthreads();
  if (tid < kP) {
    float s = 0.f;
    for (int j = 0; j < kD; ++j) s += hid[tid * kD + j] * wc2s[j];
    cvl[tid] = s;
  }
  __syncthreads();
  // x[i] += sum_j (x_j - x_i) * alpha[i][j] * cv[j]
  if (tid < kP * 3) {
    int i = tid / 3, c = tid % 3;
    float s = 0.f;
    for (int j = 0; j < kP; ++j) s += (xl[j][c] - xl[i][c]) * al[i * kP + j] * cvl[j];
    xx[(base + i) * 3 + c] += s;
  }
}

// ---------------- readout: theta = softmax_p(h @ Wr + br); out = sum_p theta * x ----------------
__global__ void k_readout(const float* __restrict__ hh, const float* __restrict__ xx,
                          const float* __restrict__ roW, const float* __restrict__ roB,
                          float* __restrict__ out) {
  int n = blockIdx.x * blockDim.x + threadIdx.x;
  if (n >= kBN) return;
  float logit[kP];
  float mx = -1e30f;
#pragma unroll
  for (int p = 0; p < kP; ++p) {
    float s = roB[0];
    for (int d = 0; d < kD; ++d) s += hh[((size_t)n * kP + p) * kD + d] * roW[d];
    logit[p] = s;
    mx = fmaxf(mx, s);
  }
  float sm = 0.f;
#pragma unroll
  for (int p = 0; p < kP; ++p) { logit[p] = __expf(logit[p] - mx); sm += logit[p]; }
  float inv = __builtin_amdgcn_rcpf(sm);
#pragma unroll
  for (int c = 0; c < 3; ++c) {
    float s = 0.f;
#pragma unroll
    for (int p = 0; p < kP; ++p) s += logit[p] * inv * xx[((size_t)n * kP + p) * 3 + c];
    out[(size_t)n * 3 + c] = s;
  }
}

// ---------------- host launcher ----------------
extern "C" void kernel_launch(void* const* d_in, const int* in_sizes, int n_in,
                              void* d_out, int out_size, void* d_ws, size_t ws_size,
                              hipStream_t stream) {
  (void)in_sizes; (void)n_in; (void)out_size; (void)ws_size;
  const float* x   = (const float*)d_in[0];
  const float* h   = (const float*)d_in[1];
  const int*   row = (const int*)d_in[2];
  const int*   col = (const int*)d_in[3];
  const float* embW = (const float*)d_in[4];
  const float* embB = (const float*)d_in[5];
  const float* temb = (const float*)d_in[6];
  const float* roW  = (const float*)d_in[7];
  const float* roB  = (const float*)d_in[8];
  // per-layer params (dict insertion order): edge(4) node(4) coord(3) q(4) k(4) v(4) at_coord(3)
  auto LPm = [&](int l, int off) { return (const float*)d_in[9 + l * 26 + off]; };

  float* w = (float*)d_ws;
  size_t o = 0;
  float* xx  = w + o; o += (size_t)kBN * kP * 3;
  float* hh  = w + o; o += (size_t)kBN * kP * kD;
  float* Fq  = w + o; o += (size_t)kBN * kEA * 6;
  float* eat = w + o; o += (size_t)kE * kEA;
  float* cnt = w + o; o += (size_t)kBN;
  float* agg = w + o; o += (size_t)kBN * kD;   // agg then xacc: contiguous for one zero pass
  float* xac = w + o; o += (size_t)kBN * 3;
  float* qb  = w + o; o += (size_t)kBN * kP * kD;
  float* kb  = w + o; o += (size_t)kBN * kP * kD;
  float* vb  = w + o; o += (size_t)kBN * kP * kD;

  const int B = 256;
  k_copy<<<(kBN * kP * 3 + B - 1) / B, B, 0, stream>>>(x, xx, kBN * kP * 3);
  k_dft<<<(kBN + B - 1) / B, B, 0, stream>>>(x, Fq);
  k_edge_attr<<<(kE + B - 1) / B, B, 0, stream>>>(Fq, row, col, eat);
  k_zero<<<(kBN + B - 1) / B, B, 0, stream>>>(cnt, kBN);
  k_cnt<<<(kE + B - 1) / B, B, 0, stream>>>(row, cnt);
  k_embed<<<(kBN * kP * kD + B - 1) / B, B, 0, stream>>>(h, embW, embB, temb, hh);

  for (int l = 0; l < kNL; ++l) {
    const float *eW1 = LPm(l, 0),  *eb1 = LPm(l, 1),  *eW2 = LPm(l, 2),  *eb2 = LPm(l, 3);
    const float *nW1 = LPm(l, 4),  *nb1 = LPm(l, 5),  *nW2 = LPm(l, 6),  *nb2 = LPm(l, 7);
    const float *cW1 = LPm(l, 8),  *cb1 = LPm(l, 9),  *cW2 = LPm(l, 10);
    const float *qW1 = LPm(l, 11), *qb1 = LPm(l, 12), *qW2 = LPm(l, 13), *qb2 = LPm(l, 14);
    const float *kW1 = LPm(l, 15), *kb1 = LPm(l, 16), *kW2 = LPm(l, 17), *kb2 = LPm(l, 18);
    const float *vW1 = LPm(l, 19), *vb1 = LPm(l, 20), *vW2 = LPm(l, 21), *vb2 = LPm(l, 22);
    const float *aW1 = LPm(l, 23), *ab1 = LPm(l, 24), *aW2 = LPm(l, 25);

    for (int t = 0; t < kP; ++t) {
      k_zero<<<(kBN * kD + kBN * 3 + B - 1) / B, B, 0, stream>>>(agg, kBN * kD + kBN * 3);
      k_edge_fused<<<kE / 16, 128, 0, stream>>>(hh, xx, row, col, eat,
                                                eW1, eb1, eW2, eb2,
                                                cW1, cb1, cW2, agg, xac, t);
      k_coord_fin<<<(kBN * 3 + B - 1) / B, B, 0, stream>>>(xx, xac, cnt, t);
      k_node_fused<<<kBN / 16, 128, 0, stream>>>(hh, agg, nW1, nb1, nW2, nb2, t);
    }
    k_mlp2<<<(kBN * kP) / 16, 128, 0, stream>>>(hh, qb, qW1, qb1, qW2, qb2);
    k_mlp2<<<(kBN * kP) / 16, 128, 0, stream>>>(hh, kb, kW1, kb1, kW2, kb2);
    k_mlp2<<<(kBN * kP) / 16, 128, 0, stream>>>(hh, vb, vW1, vb1, vW2, vb2);
    k_attn<<<kBN, 128, 0, stream>>>(hh, xx, qb, kb, vb, aW1, ab1, aW2);
  }
  k_readout<<<(kBN + B - 1) / B, B, 0, stream>>>(hh, xx, roW, roB, (float*)d_out);
}
